// DMM_22419729285741
// MI455X (gfx1250) — matrix operand order repaired and checked by reading, not compiled, not run on hardware
//
#include <hip/hip_runtime.h>
#include <stdint.h>

#define T_LEN 300
#define B_SZ  256
#define XDIM  513
#define ZDIM  16
#define HDIM  128
#define G4H   512
#define KP    544            // XDIM padded to multiple of 32
#define TBROWS (T_LEN*B_SZ)  // 76800

typedef _Float16 half_t;
typedef __attribute__((ext_vector_type(16))) half_t v16h;
typedef __attribute__((ext_vector_type(8)))  half_t v8h;
typedef __attribute__((ext_vector_type(8)))  float  v8f;
typedef __attribute__((ext_vector_type(4)))  float  f4;

union V16H { v16h v; v8h h[2]; };
union V8F  { v8f v; f4 q[2]; float f[8]; };

__device__ __forceinline__ v8f wmma16(v16h a, v16h b, v8f c) {
  // D = A(16x32 f16) * B(32x16 f16) + C(16x16 f32)
  return __builtin_amdgcn_wmma_f32_16x16x32_f16(false, a, false, b, (short)0, c, false, false);
}

// A fragment (16x32 f16) from row-major [rows x stride] buffer.
template <typename P>
__device__ __forceinline__ v16h frag_a(P base, int stride, int r0, int k0, int lane) {
  int m  = lane & 15;
  int kb = (lane < 16) ? 0 : 8;
  P p = base + (size_t)(r0 + m) * stride + k0 + kb;
  V16H a;
  a.h[0] = *(const v8h*)(p);
  a.h[1] = *(const v8h*)(p + 16);
  return a.v;
}

// B fragment (32x16 f16) from (N,K) row-major weight buffer (row = output col).
template <typename P>
__device__ __forceinline__ v16h frag_b(P base, int stride, int n0, int k0, int lane) {
  int n  = lane & 15;
  int kb = (lane < 16) ? 0 : 8;
  P p = base + (size_t)(n0 + n) * stride + k0 + kb;
  V16H b;
  b.h[0] = *(const v8h*)(p);
  b.h[1] = *(const v8h*)(p + 16);
  return b.v;
}

// fast transcendentals (hw v_exp/v_log/v_rcp paths)
__device__ __forceinline__ float fexp(float x)  { return __expf(x); }
__device__ __forceinline__ float sigf(float x)  { return __fdividef(1.f, 1.f + __expf(-x)); }
__device__ __forceinline__ float ftanh(float x) {
  float xc = fminf(fmaxf(x, -15.f), 15.f);
  float e  = __expf(2.f * xc);
  return (e - 1.f) * __fdividef(1.f, e + 1.f);
}

// ---------------------------------------------------------------- weights prep
__global__ void prep_kernel(const float* __restrict__ Wih, const float* __restrict__ Whh,
                            const float* __restrict__ bih, const float* __restrict__ bhh,
                            const float* __restrict__ Wzz, const float* __restrict__ Wim,
                            const float* __restrict__ Wil, const float* __restrict__ Wzx1,
                            const float* __restrict__ Wzx2, const float* __restrict__ Wgy,
                            half_t* wih_h, half_t* whh_h, float* bsum, half_t* wzz_h,
                            half_t* wim_h, half_t* wil_h, half_t* wzx1_h, half_t* wzx2_h,
                            half_t* wgy_h) {
  int idx = blockIdx.x * 256 + threadIdx.x;
  switch (blockIdx.y) {
    case 0: if (idx < G4H*KP) { int n = idx / KP, k = idx % KP;
              wih_h[idx] = (half_t)((k < XDIM) ? Wih[n*XDIM + k] : 0.f); } break;
    case 1: if (idx < G4H*HDIM) whh_h[idx] = (half_t)Whh[idx]; break;
    case 2: if (idx < G4H) bsum[idx] = bih[idx] + bhh[idx]; break;
    case 3: if (idx < HDIM*32) { int n = idx >> 5, k = idx & 31;
              wzz_h[idx] = (half_t)((k < ZDIM) ? Wzz[n*ZDIM + k] : 0.f); } break;
    case 4: if (idx < ZDIM*HDIM) wim_h[idx] = (half_t)Wim[idx]; break;
    case 5: if (idx < ZDIM*HDIM) wil_h[idx] = (half_t)Wil[idx]; break;
    case 6: if (idx < HDIM*32) { int n = idx >> 5, k = idx & 31;
              wzx1_h[idx] = (half_t)((k < ZDIM) ? Wzx1[n*ZDIM + k] : 0.f); } break;
    case 7: if (idx < HDIM*HDIM) wzx2_h[idx] = (half_t)Wzx2[idx]; break;
    case 8: if (idx < 528*HDIM) { int n = idx >> 7;
              wgy_h[idx] = (half_t)((n < XDIM) ? Wgy[idx] : 0.f); } break;
  }
}

// ------------------------------------------- x (B,XD,T) -> xh ((t*B+b), KP) f16
__global__ void xpose_kernel(const float* __restrict__ x, half_t* __restrict__ xh) {
  __shared__ float tile[32][33];
  int d0 = blockIdx.x * 32, t0 = blockIdx.y * 32, b = blockIdx.z;
  int tx = threadIdx.x, ty = threadIdx.y;
  int d = d0 + ty, t = t0 + tx;
  float v = 0.f;
  if (d < XDIM && t < T_LEN) v = x[((size_t)b*XDIM + d)*T_LEN + t];
  tile[ty][tx] = v;
  __syncthreads();
  int tw = t0 + ty, dw = d0 + tx;
  if (tw < T_LEN) xh[((size_t)tw*B_SZ + b)*KP + dw] = (half_t)tile[tx][ty];
}

// ------- UT(n, t*B+b) = xh @ W_ih^T + (b_ih+b_hh)
// 32x64 macro-tile per wave; each WG stages its 64-col B panel once and
// sweeps 4 row macro-tiles to amortize the staging.
__global__ void gemmU_kernel(const half_t* __restrict__ xh, const half_t* __restrict__ wih,
                             const float* __restrict__ bsum, float* __restrict__ UT) {
  extern __shared__ char smem[];
  half_t* Bs = (half_t*)smem;            // 64 x KP
  int cg = blockIdx.y;                   // column group of 64
  int tid = threadIdx.x;
  { const uint32_t* s = (const uint32_t*)(wih + (size_t)cg*64*KP);
    uint32_t* d = (uint32_t*)Bs;
    for (int i = tid; i < 64*KP/2; i += 256) d[i] = s[i]; }
  __syncthreads();
  int wave = tid >> 5, lane = tid & 31;
  int n = lane & 15, mb = (lane >> 4) << 3;
  for (int rr = 0; rr < 4; rr++) {
    int r0 = ((blockIdx.x * 4 + rr) * 8 + wave) * 32;
    V8F acc[2][4];
#pragma unroll
    for (int ct = 0; ct < 4; ct++) {
      float bv = bsum[cg*64 + ct*16 + n];
#pragma unroll
      for (int e = 0; e < 8; e++) { acc[0][ct].f[e] = bv; acc[1][ct].f[e] = bv; }
    }
    for (int ks = 0; ks < KP/32; ks++) {
      v16h a0 = frag_a(xh, KP, r0,      ks*32, lane);
      v16h a1 = frag_a(xh, KP, r0 + 16, ks*32, lane);
      v16h bf[4];
#pragma unroll
      for (int ct = 0; ct < 4; ct++) bf[ct] = frag_b(Bs, KP, ct*16, ks*32, lane);
#pragma unroll
      for (int ct = 0; ct < 4; ct++) {
        acc[0][ct].v = wmma16(a0, bf[ct], acc[0][ct].v);
        acc[1][ct].v = wmma16(a1, bf[ct], acc[1][ct].v);
      }
    }
#pragma unroll
    for (int h2 = 0; h2 < 2; h2++)
#pragma unroll
      for (int ct = 0; ct < 4; ct++) {
        float* out = UT + (size_t)(cg*64 + ct*16 + n)*TBROWS + r0 + h2*16 + mb;
        *(f4*)out = acc[h2][ct].q[0];
        *(f4*)(out + 4) = acc[h2][ct].q[1];
      }
  }
}

// ------------- LSTM over reversed time, single persistent workgroup (16 waves)
// Each wave owns 4 tasks; each task = 32 batch rows x 16 hidden cols of all 4
// gates (i/f/g/o meet in registers, c-state lives in VGPRs for all 300 steps).
__global__ void lstm_kernel(const float* __restrict__ UT, const half_t* __restrict__ whh,
                            float* __restrict__ g) {
  extern __shared__ char smem[];
  half_t* Ws   = (half_t*)smem;                    // 512 x 128
  half_t* hbuf = Ws + G4H*HDIM;                    // 2 x 256 x 128 (double buffer)
  int tid = threadIdx.x, wave = tid >> 5, lane = tid & 31;
  { const uint32_t* s = (const uint32_t*)whh; uint32_t* d = (uint32_t*)Ws;
    for (int i = tid; i < G4H*HDIM/2; i += 512) d[i] = s[i];
    uint32_t* z = (uint32_t*)hbuf;
    for (int i = tid; i < 2*B_SZ*HDIM/2; i += 512) z[i] = 0u; }
  __syncthreads();
  int n = lane & 15, mb = (lane >> 4) << 3;
  float creg[4][2][8];
#pragma unroll
  for (int q = 0; q < 4; q++)
#pragma unroll
    for (int h2 = 0; h2 < 2; h2++)
#pragma unroll
      for (int e = 0; e < 8; e++) creg[q][h2][e] = 0.f;
  int sel = 0;
  for (int s = 0; s < T_LEN; s++) {
    int t = T_LEN - 1 - s;
    half_t* hr = hbuf + sel * B_SZ*HDIM;
    half_t* hw = hbuf + (sel ^ 1) * B_SZ*HDIM;
#pragma unroll
    for (int q = 0; q < 4; q++) {
      int id = wave*4 + q;                 // 0..63
      int b0 = (id >> 3) * 32, j0 = (id & 7) * 16, j = j0 + n;
      V8F acc[2][4];
#pragma unroll
      for (int h2 = 0; h2 < 2; h2++)
#pragma unroll
        for (int gg = 0; gg < 4; gg++) {   // i,f,g,o
          const float* up = UT + (size_t)(gg*HDIM + j)*TBROWS + (size_t)t*B_SZ + b0 + h2*16 + mb;
          acc[h2][gg].q[0] = *(const f4*)up;
          acc[h2][gg].q[1] = *(const f4*)(up + 4);
        }
#pragma unroll
      for (int ks = 0; ks < 4; ks++) {
        v16h a0 = frag_a(hr, HDIM, b0,      ks*32, lane);
        v16h a1 = frag_a(hr, HDIM, b0 + 16, ks*32, lane);
        v16h bf[4];
#pragma unroll
        for (int gg = 0; gg < 4; gg++) bf[gg] = frag_b(Ws, HDIM, gg*HDIM + j0, ks*32, lane);
#pragma unroll
        for (int gg = 0; gg < 4; gg++) {
          acc[0][gg].v = wmma16(a0, bf[gg], acc[0][gg].v);
          acc[1][gg].v = wmma16(a1, bf[gg], acc[1][gg].v);
        }
      }
#pragma unroll
      for (int h2 = 0; h2 < 2; h2++) {
        float hv[8];
#pragma unroll
        for (int e = 0; e < 8; e++) {
          float iv = sigf(acc[h2][0].f[e]), fv = sigf(acc[h2][1].f[e]);
          float gv = ftanh(acc[h2][2].f[e]), ov = sigf(acc[h2][3].f[e]);
          float c = fv * creg[q][h2][e] + iv * gv;
          creg[q][h2][e] = c;
          hv[e] = ov * ftanh(c);
        }
        int br = b0 + h2*16 + mb;
#pragma unroll
        for (int e = 0; e < 8; e++) hw[(size_t)(br + e)*HDIM + j] = (half_t)hv[e];
        float* gp = g + ((size_t)t*HDIM + j)*B_SZ + br;   // layout (T,H,B)
        f4 lo = {hv[0], hv[1], hv[2], hv[3]};
        f4 hi = {hv[4], hv[5], hv[6], hv[7]};
        *(f4*)gp = lo; *(f4*)(gp + 4) = hi;
      }
    }
    __syncthreads();
    sel ^= 1;
  }
}

// ------------------------ inference scan (z recurrence), single workgroup
__global__ void inf_kernel(const float* __restrict__ g, const float* __restrict__ eps,
                           const half_t* __restrict__ wzz, const float* __restrict__ bzz,
                           const half_t* __restrict__ wim, const float* __restrict__ bim,
                           const half_t* __restrict__ wil, const float* __restrict__ bil,
                           float* __restrict__ zm, float* __restrict__ zl,
                           float* __restrict__ zbuf) {
  extern __shared__ char smem[];
  half_t* zS   = (half_t*)smem;          // 256 x 32 (K-padded z, cols 16..31 = 0)
  half_t* gtS  = zS + B_SZ*32;           // 256 x 128
  half_t* WzzS = gtS + B_SZ*HDIM;        // 128 x 32
  half_t* WimS = WzzS + HDIM*32;         // 16 x 128
  half_t* WilS = WimS + ZDIM*HDIM;       // 16 x 128
  int tid = threadIdx.x, wave = tid >> 5, lane = tid & 31;
  { uint32_t* d = (uint32_t*)zS;
    for (int i = tid; i < B_SZ*32/2; i += 1024) d[i] = 0u;
    d = (uint32_t*)WzzS; const uint32_t* s = (const uint32_t*)wzz;
    for (int i = tid; i < HDIM*32/2; i += 1024) d[i] = s[i];
    d = (uint32_t*)WimS; s = (const uint32_t*)wim;
    for (int i = tid; i < ZDIM*HDIM/2; i += 1024) d[i] = s[i];
    d = (uint32_t*)WilS; s = (const uint32_t*)wil;
    for (int i = tid; i < ZDIM*HDIM/2; i += 1024) d[i] = s[i]; }
  __syncthreads();
  int n = lane & 15, mb = (lane >> 4) << 3;
  for (int t = 0; t < T_LEN; t++) {
    // phase 1: gt = (tanh(z @ Wzz^T + bzz) + g_t) / 2   -> gtS
#pragma unroll
    for (int q = 0; q < 4; q++) {
      int id = wave*4 + q;
      int b0 = (id >> 3) * 16, j0 = (id & 7) * 16, j = j0 + n;
      V8F acc;
#pragma unroll
      for (int e = 0; e < 8; e++) acc.f[e] = 0.f;
      v16h a  = frag_a(zS, 32, b0, 0, lane);
      v16h bm = frag_b(WzzS, 32, j0, 0, lane);
      acc.v = wmma16(a, bm, acc.v);
      float bj = bzz[j];
      const float* gp = g + ((size_t)t*HDIM + j)*B_SZ + b0 + mb;
      f4 g0 = *(const f4*)gp;
      f4 g1 = *(const f4*)(gp + 4);
#pragma unroll
      for (int e = 0; e < 8; e++) {
        float gv = (e < 4) ? g0[e] : g1[e - 4];
        float gt = 0.5f * (ftanh(acc.f[e] + bj) + gv);
        gtS[(size_t)(b0 + mb + e)*HDIM + j] = (half_t)gt;
      }
    }
    __syncthreads();
    // phase 2: mean/logvar = gt @ W^T + b ; z = mean + eps*exp(0.5*logvar)
    if (wave < 16) {
      int b0 = wave * 16;
      V8F am, al;
#pragma unroll
      for (int e = 0; e < 8; e++) { am.f[e] = 0.f; al.f[e] = 0.f; }
#pragma unroll
      for (int ks = 0; ks < 4; ks++) {
        v16h a   = frag_a(gtS, HDIM, b0, ks*32, lane);
        v16h b0f = frag_b(WimS, HDIM, 0, ks*32, lane);
        v16h b1f = frag_b(WilS, HDIM, 0, ks*32, lane);
        am.v = wmma16(a, b0f, am.v);
        al.v = wmma16(a, b1f, al.v);
      }
      float bmv = bim[n], blv = bil[n];
#pragma unroll
      for (int e = 0; e < 8; e++) {
        int br = b0 + mb + e;
        size_t idx = ((size_t)t*B_SZ + br)*ZDIM + n;
        float mean = am.f[e] + bmv;
        float lv   = al.f[e] + blv;
        float z = mean + eps[idx] * fexp(0.5f * lv);
        zm[idx] = mean; zl[idx] = lv; zbuf[idx] = z;
        zS[(size_t)br*32 + n] = (half_t)z;
      }
    }
    __syncthreads();
  }
}

// ---------------- generation: MLPs + y = exp(h2 @ Wgy^T); rows = 16 consecutive t
__global__ void gen_kernel(const float* __restrict__ zbuf,
                           const float* __restrict__ Wg1, const float* __restrict__ bg1,
                           const float* __restrict__ Wg2, const float* __restrict__ bg2,
                           const float* __restrict__ Wp1, const float* __restrict__ bp1,
                           const float* __restrict__ Wp2, const float* __restrict__ bp2,
                           const float* __restrict__ Wpm, const float* __restrict__ bpm,
                           const float* __restrict__ Wpl, const float* __restrict__ bpl,
                           const half_t* __restrict__ wzx1, const float* __restrict__ bzx1,
                           const half_t* __restrict__ wzx2, const float* __restrict__ bzx2,
                           const half_t* __restrict__ wgy, const float* __restrict__ bgy,
                           float* __restrict__ y_out, float* __restrict__ zmp_out,
                           float* __restrict__ zlp_out, float* __restrict__ zt_out) {
  extern __shared__ char smem[];
  half_t* zS  = (half_t*)smem;       // 128 x 32
  half_t* h1S = zS + 128*32;         // 128 x 128
  half_t* h2S = h1S + 128*HDIM;      // 128 x 128
  int bb = blockIdx.y;
  int t0 = blockIdx.x * 128;
  int tid = threadIdx.x, wave = tid >> 5, lane = tid & 31;
  for (int idx = tid; idx < 128*32; idx += 256) {
    int i = idx >> 5, k = idx & 31, t = t0 + i;
    float v = (k < ZDIM && t < T_LEN) ? zbuf[((size_t)t*B_SZ + bb)*ZDIM + k] : 0.f;
    zS[idx] = (half_t)v;
  }
  __syncthreads();
  // small ZDxZD MLPs on VALU (one thread per row)
  if (tid < 128) {
    int t = t0 + tid;
    if (t < T_LEN) {
      float z[ZDIM], a1[ZDIM], gate[ZDIM], p1[ZDIM], zprop[ZDIM];
#pragma unroll
      for (int k = 0; k < ZDIM; k++) z[k] = (float)zS[tid*32 + k];
#pragma unroll
      for (int o = 0; o < ZDIM; o++) {
        float s = bg1[o];
        for (int k = 0; k < ZDIM; k++) s += z[k] * Wg1[o*ZDIM + k];
        a1[o] = fmaxf(s, 0.f);
      }
#pragma unroll
      for (int o = 0; o < ZDIM; o++) {
        float s = bg2[o];
        for (int k = 0; k < ZDIM; k++) s += a1[k] * Wg2[o*ZDIM + k];
        gate[o] = sigf(s);
      }
#pragma unroll
      for (int o = 0; o < ZDIM; o++) {
        float s = bp1[o];
        for (int k = 0; k < ZDIM; k++) s += z[k] * Wp1[o*ZDIM + k];
        p1[o] = fmaxf(s, 0.f);
      }
#pragma unroll
      for (int o = 0; o < ZDIM; o++) {
        float s = bp2[o];
        for (int k = 0; k < ZDIM; k++) s += p1[k] * Wp2[o*ZDIM + k];
        zprop[o] = s;
      }
      size_t base = ((size_t)t*B_SZ + bb)*ZDIM;
#pragma unroll
      for (int o = 0; o < ZDIM; o++) {
        float zlin = bpm[o];
        for (int k = 0; k < ZDIM; k++) zlin += z[k] * Wpm[o*ZDIM + k];
        zmp_out[base + o] = (1.f - gate[o]) * zlin + gate[o] * zprop[o];
      }
#pragma unroll
      for (int o = 0; o < ZDIM; o++) {
        float s = bpl[o];
        for (int k = 0; k < ZDIM; k++) s += fmaxf(zprop[k], 0.f) * Wpl[o*ZDIM + k];
        float sp = (s > 20.f) ? s : __logf(1.f + __expf(s));
        zlp_out[base + o] = __logf(sp);
      }
    }
  }
  // z_out transpose: (T,B,ZD) -> (B,ZD,T), coalesced along t
  for (int idx = tid; idx < ZDIM*128; idx += 256) {
    int o = idx >> 7, i = idx & 127, t = t0 + i;
    if (t < T_LEN)
      zt_out[((size_t)bb*ZDIM + o)*T_LEN + t] = zbuf[((size_t)t*B_SZ + bb)*ZDIM + o];
  }
  int n = lane & 15, mb = (lane >> 4) << 3;
  // h1 = tanh(z @ Wzx1^T + b)
#pragma unroll
  for (int q = 0; q < 8; q++) {
    int id = wave*8 + q, rt = id >> 3, ct = id & 7;
    V8F acc;
#pragma unroll
    for (int e = 0; e < 8; e++) acc.f[e] = 0.f;
    v16h a  = frag_a(zS, 32, rt*16, 0, lane);
    v16h bm = frag_b(wzx1, 32, ct*16, 0, lane);
    acc.v = wmma16(a, bm, acc.v);
    float bv = bzx1[ct*16 + n];
#pragma unroll
    for (int e = 0; e < 8; e++)
      h1S[(size_t)(rt*16 + mb + e)*HDIM + ct*16 + n] = (half_t)ftanh(acc.f[e] + bv);
  }
  __syncthreads();
  // h2 = tanh(h1 @ Wzx2^T + b)
#pragma unroll
  for (int q = 0; q < 8; q++) {
    int id = wave*8 + q, rt = id >> 3, ct = id & 7;
    V8F acc;
#pragma unroll
    for (int e = 0; e < 8; e++) acc.f[e] = 0.f;
    v16h bf[4];
#pragma unroll
    for (int ks = 0; ks < 4; ks++) bf[ks] = frag_b(wzx2, HDIM, ct*16, ks*32, lane);
#pragma unroll
    for (int ks = 0; ks < 4; ks++) {
      v16h a = frag_a(h1S, HDIM, rt*16, ks*32, lane);
      acc.v = wmma16(a, bf[ks], acc.v);
    }
    float bv = bzx2[ct*16 + n];
#pragma unroll
    for (int e = 0; e < 8; e++)
      h2S[(size_t)(rt*16 + mb + e)*HDIM + ct*16 + n] = (half_t)ftanh(acc.f[e] + bv);
  }
  __syncthreads();
  // y = exp(h2 @ Wgy^T + b), stored transposed (B, XD, T) with contiguous-t b128 stores
  int rt = wave;
  v16h afr[4];
#pragma unroll
  for (int ks = 0; ks < 4; ks++) afr[ks] = frag_a(h2S, HDIM, rt*16, ks*32, lane);
  int tb = t0 + rt*16 + mb;
  for (int ct = 0; ct < 33; ct++) {
    V8F acc;
#pragma unroll
    for (int e = 0; e < 8; e++) acc.f[e] = 0.f;
    v16h bf[4];
#pragma unroll
    for (int ks = 0; ks < 4; ks++) bf[ks] = frag_b(wgy, HDIM, ct*16, ks*32, lane);
#pragma unroll
    for (int ks = 0; ks < 4; ks++) acc.v = wmma16(afr[ks], bf[ks], acc.v);
    int d = ct*16 + n;
    if (d < XDIM && tb < T_LEN) {
      float bv = bgy[d];
      float* yp = y_out + ((size_t)bb*XDIM + d)*T_LEN + tb;
      if (tb + 7 < T_LEN) {
        f4 lo = {fexp(acc.f[0]+bv), fexp(acc.f[1]+bv), fexp(acc.f[2]+bv), fexp(acc.f[3]+bv)};
        f4 hi = {fexp(acc.f[4]+bv), fexp(acc.f[5]+bv), fexp(acc.f[6]+bv), fexp(acc.f[7]+bv)};
        *(f4*)yp = lo; *(f4*)(yp + 4) = hi;
      } else {
#pragma unroll
        for (int e = 0; e < 8; e++)
          if (tb + e < T_LEN) yp[e] = fexp(acc.f[e] + bv);
      }
    }
  }
}

extern "C" void kernel_launch(void* const* d_in, const int* in_sizes, int n_in,
                              void* d_out, int out_size, void* d_ws, size_t ws_size,
                              hipStream_t stream) {
  (void)in_sizes; (void)n_in; (void)out_size; (void)ws_size;
  const float* x    = (const float*)d_in[0];
  const float* eps  = (const float*)d_in[1];
  const float* W_ih = (const float*)d_in[2];
  const float* W_hh = (const float*)d_in[3];
  const float* b_ih = (const float*)d_in[4];
  const float* b_hh = (const float*)d_in[5];
  const float* W_zz = (const float*)d_in[6];
  const float* b_zz = (const float*)d_in[7];
  const float* W_im = (const float*)d_in[8];
  const float* b_im = (const float*)d_in[9];
  const float* W_il = (const float*)d_in[10];
  const float* b_il = (const float*)d_in[11];
  const float* W_g1 = (const float*)d_in[12];
  const float* b_g1 = (const float*)d_in[13];
  const float* W_g2 = (const float*)d_in[14];
  const float* b_g2 = (const float*)d_in[15];
  const float* W_p1 = (const float*)d_in[16];
  const float* b_p1 = (const float*)d_in[17];
  const float* W_p2 = (const float*)d_in[18];
  const float* b_p2 = (const float*)d_in[19];
  const float* W_pm = (const float*)d_in[20];
  const float* b_pm = (const float*)d_in[21];
  const float* W_pl = (const float*)d_in[22];
  const float* b_pl = (const float*)d_in[23];
  const float* W_zx1= (const float*)d_in[24];
  const float* b_zx1= (const float*)d_in[25];
  const float* W_zx2= (const float*)d_in[26];
  const float* b_zx2= (const float*)d_in[27];
  const float* W_gy = (const float*)d_in[28];
  const float* b_gy = (const float*)d_in[29];

  char* ws = (char*)d_ws;
  size_t off = 0;
  auto carve = [&](size_t bytes) -> void* {
    void* p = ws + off;
    off = (off + bytes + 255) & ~(size_t)255;
    return p;
  };
  half_t* xh    = (half_t*)carve((size_t)TBROWS*KP*2);
  float*  UT    = (float*) carve((size_t)G4H*TBROWS*4);
  float*  gbuf  = (float*) carve((size_t)T_LEN*HDIM*B_SZ*4);
  float*  zbuf  = (float*) carve((size_t)T_LEN*B_SZ*ZDIM*4);
  half_t* wih_h = (half_t*)carve((size_t)G4H*KP*2);
  half_t* whh_h = (half_t*)carve((size_t)G4H*HDIM*2);
  float*  bsum  = (float*) carve((size_t)G4H*4);
  half_t* wzz_h = (half_t*)carve((size_t)HDIM*32*2);
  half_t* wim_h = (half_t*)carve((size_t)ZDIM*HDIM*2);
  half_t* wil_h = (half_t*)carve((size_t)ZDIM*HDIM*2);
  half_t* wzx1_h= (half_t*)carve((size_t)HDIM*32*2);
  half_t* wzx2_h= (half_t*)carve((size_t)HDIM*HDIM*2);
  half_t* wgy_h = (half_t*)carve((size_t)528*HDIM*2);

  float* y_out   = (float*)d_out;
  float* zm_out  = y_out  + (size_t)B_SZ*XDIM*T_LEN;
  float* zl_out  = zm_out + (size_t)T_LEN*B_SZ*ZDIM;
  float* zmp_out = zl_out + (size_t)T_LEN*B_SZ*ZDIM;
  float* zlp_out = zmp_out+ (size_t)T_LEN*B_SZ*ZDIM;
  float* zt_out  = zlp_out+ (size_t)T_LEN*B_SZ*ZDIM;

  prep_kernel<<<dim3(1088, 9), 256, 0, stream>>>(W_ih, W_hh, b_ih, b_hh, W_zz, W_im, W_il,
      W_zx1, W_zx2, W_gy, wih_h, whh_h, bsum, wzz_h, wim_h, wil_h, wzx1_h, wzx2_h, wgy_h);
  xpose_kernel<<<dim3(17, 10, 256), dim3(32, 32), 0, stream>>>(x, xh);
  gemmU_kernel<<<dim3(75, 8), 256, 64*KP*2, stream>>>(xh, wih_h, bsum, UT);
  lstm_kernel<<<1, 512, (G4H*HDIM + 2*B_SZ*HDIM)*2, stream>>>(UT, whh_h, gbuf);
  inf_kernel<<<1, 1024, (B_SZ*32 + B_SZ*HDIM + HDIM*32 + 2*ZDIM*HDIM)*2, stream>>>(
      gbuf, eps, wzz_h, b_zz, wim_h, b_im, wil_h, b_il, zm_out, zl_out, zbuf);
  gen_kernel<<<dim3(3, B_SZ), 256, (128*32 + 2*128*HDIM)*2, stream>>>(zbuf,
      W_g1, b_g1, W_g2, b_g2, W_p1, b_p1, W_p2, b_p2, W_pm, b_pm, W_pl, b_pl,
      wzx1_h, b_zx1, wzx2_h, b_zx2, wgy_h, b_gy, y_out, zmp_out, zlp_out, zt_out);
}
